// MeshGNN_73220602462590
// MI455X (gfx1250) — matrix operand order repaired and checked by reading, not compiled, not run on hardware
//
#include <hip/hip_runtime.h>
#include <hip/hip_bf16.h>

// ---------------------------------------------------------------------------
// MeshGNN on MI455X (gfx1250): bf16 WMMA GEMMs, L2-resident atomic scatter,
// software-pipelined K-loops, 16-byte vectorized LDS staging.
//   N=50000 nodes, E=600000 edges, IN=16, ED=4, H=128, L=4, OUT=3
// ---------------------------------------------------------------------------

#define N_NODES 50000
#define N_EDGES 600000
#define DIM_IN  16
#define DIM_ED  4
#define DIM_H   128
#define N_LAYERS 4
#define DIM_OUT 3
#define KP1 160   // H+ED=132 padded to multiple of 32
#define KU  256   // 2H for update MLP input

typedef __attribute__((ext_vector_type(16))) __bf16 bf16x16;
typedef __attribute__((ext_vector_type(8)))  __bf16 bf16x8;
typedef __attribute__((ext_vector_type(8)))  float  f32x8;
typedef __attribute__((ext_vector_type(4)))  float  f32x4;

// fp32 -> bf16 round-to-nearest-even (bit level)
__device__ __forceinline__ __bf16 f2bf(float f) {
    union { float f; unsigned u; } v; v.f = f;
    unsigned r = (v.u + 0x7FFFu + ((v.u >> 16) & 1u)) >> 16;
    union { unsigned short s; __bf16 b; } o; o.s = (unsigned short)r;
    return o.b;
}

__device__ __forceinline__ bf16x8 f2bf8(f32x4 a, f32x4 b) {
    bf16x8 r;
    r[0] = f2bf(a[0]); r[1] = f2bf(a[1]); r[2] = f2bf(a[2]); r[3] = f2bf(a[3]);
    r[4] = f2bf(b[0]); r[5] = f2bf(b[1]); r[6] = f2bf(b[2]); r[7] = f2bf(b[3]);
    return r;
}

// A-fragment per ISA 16-bit A 16x32 layout:
//   lane half=l>>4; elements 0..7 = K(half*8 + 0..7), 8..15 = K(16 + half*8 + 0..7)
// p points at &row[half*8 + kb*32]; two contiguous 16B LDS loads.
__device__ __forceinline__ bf16x16 load_a16(const __bf16* p) {
    bf16x8 lo = *(const bf16x8*)(p);
    bf16x8 hi = *(const bf16x8*)(p + 16);
    return __builtin_shufflevector(lo, hi, 0,1,2,3,4,5,6,7,8,9,10,11,12,13,14,15);
}

// One 16x16 output tile, K = KCHUNKS*32, A from LDS (row stride ASTRIDE elems),
// B from packed transposed bf16 weights (wT[n][k], k contiguous).
// Software-pipelined: chunk k+1 fragment loads are issued before WMMA on chunk k.
template<int KCHUNKS, int ASTRIDE>
__device__ __forceinline__ f32x8 gemm_tile(const __bf16* Atile, const __bf16* BcolPtr, int lane) {
    const int half = lane >> 4;
    const __bf16* ap = Atile + (lane & 15) * ASTRIDE + half * 8;
    const __bf16* bp = BcolPtr + half * 8;    // B frag: K(half*8+0..7), K(16+half*8+0..7) of column n
    f32x8 c = {};
    bf16x16 a = load_a16(ap);
    bf16x16 b = load_a16(bp);
#pragma unroll
    for (int kb = 0; kb < KCHUNKS - 1; kb++) {
        bf16x16 an = load_a16(ap + (kb + 1) * 32);
        bf16x16 bn = load_a16(bp + (kb + 1) * 32);
        c = __builtin_amdgcn_wmma_f32_16x16x32_bf16(false, a, false, b,
                                                    (short)0, c, false, false);
        a = an; b = bn;
    }
    c = __builtin_amdgcn_wmma_f32_16x16x32_bf16(false, a, false, b,
                                                (short)0, c, false, false);
    return c;
}

// ---------------------------------------------------------------------------
// Utility kernels
// ---------------------------------------------------------------------------
__global__ void zero_f32x4(f32x4* p, int n4) {
    int i = blockIdx.x * blockDim.x + threadIdx.x;
    if (i < n4) p[i] = (f32x4){0.f, 0.f, 0.f, 0.f};
}

__global__ void zero_f32(float* p, int n) {
    int i = blockIdx.x * blockDim.x + threadIdx.x;
    if (i < n) p[i] = 0.f;
}

__global__ void count_deg(const int* ei, float* cnt, int E) {
    int e = blockIdx.x * blockDim.x + threadIdx.x;
    if (e < E) atomicAdd(&cnt[ei[E + e]], 1.0f);
}

// Pack fp32 weight [batch][K][Nn] -> bf16 transposed+padded [batch][Nn][Kpad]
__global__ void pack_wT(const float* __restrict__ w, __bf16* __restrict__ out,
                        int batch, int K, int Nn, int Kpad) {
    int i = blockIdx.x * blockDim.x + threadIdx.x;
    int total = batch * Nn * Kpad;
    if (i >= total) return;
    int k = i % Kpad; int t = i / Kpad; int n = t % Nn; int b = t / Nn;
    float v = (k < K) ? w[((size_t)b * K + k) * Nn + n] : 0.f;
    out[i] = f2bf(v);
}

// h = relu(x @ W_in + b); also keep bf16 copy for WMMA / gathers
__global__ void lin_in_kernel(const float* __restrict__ x, const float* __restrict__ w,
                              const float* __restrict__ b,
                              float* __restrict__ h, __bf16* __restrict__ hb) {
    int i = blockIdx.x * blockDim.x + threadIdx.x;
    if (i >= N_NODES * DIM_H) return;
    int n = i >> 7, j = i & 127;
    float acc = b[j];
#pragma unroll
    for (int k = 0; k < DIM_IN; k++) acc = fmaf(x[n * DIM_IN + k], w[k * DIM_H + j], acc);
    acc = fmaxf(acc, 0.f);
    h[i] = acc;
    hb[i] = f2bf(acc);
}

// ---------------------------------------------------------------------------
// Per-edge message MLP: m = relu(relu([h[src],ea] @ W1 + b1) @ W2 + b2),
// then atomic scatter-add into agg[dst].  16 edges/block, 8 waves (N-tiles).
// ---------------------------------------------------------------------------
__global__ __launch_bounds__(256)
void msg_kernel(const __bf16* __restrict__ hb, const int* __restrict__ ei,
                const float* __restrict__ ea,
                const __bf16* __restrict__ w1p, const float* __restrict__ b1,
                const __bf16* __restrict__ w2p, const float* __restrict__ b2,
                float* __restrict__ agg) {
    __shared__ __align__(16) __bf16 At[16 * KP1];
    __shared__ __align__(16) __bf16 Mt[16 * DIM_H];
    __shared__ int s_src[16];
    __shared__ int s_dst[16];

    const int t  = threadIdx.x;
    const int e0 = blockIdx.x * 16;

    if (t < 16) {
        s_src[t] = ei[e0 + t];
        s_dst[t] = ei[N_EDGES + e0 + t];
    }
    __syncthreads();

    // Vectorized staging: 256 threads = 16 rows x 16 chunks.
    {
        const int r  = t >> 4;         // edge row in tile
        const int ch = t & 15;         // 8-element chunk within row
        // h[src] gather: 16B per thread (bf16x8)
        *(bf16x8*)&At[r * KP1 + ch * 8] =
            *(const bf16x8*)&hb[(size_t)s_src[r] * DIM_H + ch * 8];
        // tail: edge_attr (4) + zero pad (28), 2 elements per thread
        const int k0 = DIM_H + ch * 2;
        const int e  = e0 + r;
        At[r * KP1 + k0]     = (k0     < DIM_H + DIM_ED) ? f2bf(ea[(size_t)e * DIM_ED + (k0 - DIM_H)])     : f2bf(0.f);
        At[r * KP1 + k0 + 1] = (k0 + 1 < DIM_H + DIM_ED) ? f2bf(ea[(size_t)e * DIM_ED + (k0 + 1 - DIM_H)]) : f2bf(0.f);
    }
    __syncthreads();

    const int lane  = t & 31;
    const int nbase = (t >> 5) * 16;           // wave -> N-tile
    const int col   = lane & 15;
    const int half  = lane >> 4;

    const __bf16* b1col = w1p + (size_t)(nbase + col) * KP1;
    const __bf16* b2col = w2p + (size_t)(nbase + col) * DIM_H;
    __builtin_prefetch(b1col, 0, 1);           // global_prefetch_b8
    __builtin_prefetch(b2col, 0, 1);

    // GEMM1: [16,160] x [160,16] -> relu -> Mt (bf16)
    f32x8 c = gemm_tile<KP1 / 32, KP1>(At, b1col, lane);
    float bias1 = b1[nbase + col];
#pragma unroll
    for (int r = 0; r < 8; r++)
        Mt[(half * 8 + r) * DIM_H + nbase + col] = f2bf(fmaxf(c[r] + bias1, 0.f));
    __syncthreads();

    // GEMM2: [16,128] x [128,16] -> relu -> atomic scatter to agg[dst]
    f32x8 c2 = gemm_tile<DIM_H / 32, DIM_H>(Mt, b2col, lane);
    float bias2 = b2[nbase + col];
#pragma unroll
    for (int r = 0; r < 8; r++) {
        float v = fmaxf(c2[r] + bias2, 0.f);
        int row = half * 8 + r;
        atomicAdd(&agg[(size_t)s_dst[row] * DIM_H + nbase + col], v);
    }
}

// ---------------------------------------------------------------------------
// Per-node update MLP: u = relu([h, agg/cnt] @ U1 + b1) @ U2 + b2;
// h = relu(u + h).  16 nodes/block, 8 waves.
// ---------------------------------------------------------------------------
__global__ __launch_bounds__(256)
void upd_kernel(float* __restrict__ h, __bf16* __restrict__ hb,
                const float* __restrict__ agg, const float* __restrict__ cnt,
                const __bf16* __restrict__ w1p, const float* __restrict__ b1,
                const __bf16* __restrict__ w2p, const float* __restrict__ b2) {
    __shared__ __align__(16) __bf16 At[16 * KU];
    __shared__ __align__(16) __bf16 Ut[16 * DIM_H];

    const int t  = threadIdx.x;
    const int n0 = blockIdx.x * 16;

    // Vectorized staging: 256 threads = 16 rows x 16 chunks of 8 elements.
    {
        const int r    = t >> 4;
        const int ch   = t & 15;
        const int node = n0 + r;
        // h-part straight from bf16 mirror (16B copy)
        *(bf16x8*)&At[r * KU + ch * 8] =
            *(const bf16x8*)&hb[(size_t)node * DIM_H + ch * 8];
        // agg-part: mean = agg / max(cnt,1), f32 -> bf16
        const float rc = 1.0f / fmaxf(cnt[node], 1.f);
        const float* ap = &agg[(size_t)node * DIM_H + ch * 8];
        f32x4 a0 = *(const f32x4*)(ap);
        f32x4 a1 = *(const f32x4*)(ap + 4);
        a0 *= rc; a1 *= rc;
        *(bf16x8*)&At[r * KU + DIM_H + ch * 8] = f2bf8(a0, a1);
    }
    __syncthreads();

    const int lane  = t & 31;
    const int nbase = (t >> 5) * 16;
    const int col   = lane & 15;
    const int half  = lane >> 4;

    const __bf16* b1col = w1p + (size_t)(nbase + col) * KU;
    const __bf16* b2col = w2p + (size_t)(nbase + col) * DIM_H;
    __builtin_prefetch(b1col, 0, 1);
    __builtin_prefetch(b2col, 0, 1);

    f32x8 c = gemm_tile<KU / 32, KU>(At, b1col, lane);
    float bias1 = b1[nbase + col];
#pragma unroll
    for (int r = 0; r < 8; r++)
        Ut[(half * 8 + r) * DIM_H + nbase + col] = f2bf(fmaxf(c[r] + bias1, 0.f));
    __syncthreads();

    f32x8 c2 = gemm_tile<DIM_H / 32, DIM_H>(Ut, b2col, lane);
    float bias2 = b2[nbase + col];
#pragma unroll
    for (int r = 0; r < 8; r++) {
        int row = half * 8 + r;
        size_t o = (size_t)(n0 + row) * DIM_H + nbase + col;
        // residual from fp32 h (each element read+written by exactly one lane)
        float v = fmaxf(c2[r] + bias2 + h[o], 0.f);
        h[o]  = v;
        hb[o] = f2bf(v);
    }
}

// ---------------------------------------------------------------------------
// Head: o = relu(h @ Hw1 + b1) [WMMA]; out = relu(o @ Hw2 + b2) [OUT=3, VALU]
// ---------------------------------------------------------------------------
__global__ __launch_bounds__(256)
void head_kernel(const __bf16* __restrict__ hb,
                 const __bf16* __restrict__ w1p, const float* __restrict__ b1,
                 const float* __restrict__ w2, const float* __restrict__ b2,
                 float* __restrict__ out) {
    __shared__ __align__(16) __bf16 At[16 * DIM_H];
    __shared__ float Ot[16 * DIM_H];

    const int t  = threadIdx.x;
    const int n0 = blockIdx.x * 16;

    // Flat 16B-per-thread copy: 256 threads x 16B = 4KB tile
    *(bf16x8*)&At[t * 8] = *(const bf16x8*)&hb[(size_t)n0 * DIM_H + t * 8];
    __syncthreads();

    const int lane  = t & 31;
    const int nbase = (t >> 5) * 16;
    const int col   = lane & 15;
    const int half  = lane >> 4;

    const __bf16* bcol = w1p + (size_t)(nbase + col) * DIM_H;
    __builtin_prefetch(bcol, 0, 1);
    f32x8 c = gemm_tile<DIM_H / 32, DIM_H>(At, bcol, lane);
    float bias1 = b1[nbase + col];
#pragma unroll
    for (int r = 0; r < 8; r++)
        Ot[(half * 8 + r) * DIM_H + nbase + col] = fmaxf(c[r] + bias1, 0.f);
    __syncthreads();

    if (t < 16 * DIM_OUT) {
        int r = t / DIM_OUT, j = t - r * DIM_OUT;
        float acc = b2[j];
#pragma unroll 8
        for (int k = 0; k < DIM_H; k++)
            acc = fmaf(Ot[r * DIM_H + k], w2[k * DIM_OUT + j], acc);
        out[(size_t)(n0 + r) * DIM_OUT + j] = fmaxf(acc, 0.f);
    }
}

// ---------------------------------------------------------------------------
extern "C" void kernel_launch(void* const* d_in, const int* in_sizes, int n_in,
                              void* d_out, int out_size, void* d_ws, size_t ws_size,
                              hipStream_t stream) {
    (void)in_sizes; (void)n_in; (void)out_size; (void)ws_size;

    const float* x       = (const float*)d_in[0];
    const int*   ei      = (const int*)  d_in[1];
    const float* ea      = (const float*)d_in[2];
    const float* lin_w   = (const float*)d_in[3];
    const float* lin_b   = (const float*)d_in[4];
    const float* msg_w1  = (const float*)d_in[5];
    const float* msg_b1  = (const float*)d_in[6];
    const float* msg_w2  = (const float*)d_in[7];
    const float* msg_b2  = (const float*)d_in[8];
    const float* upd_w1  = (const float*)d_in[9];
    const float* upd_b1  = (const float*)d_in[10];
    const float* upd_w2  = (const float*)d_in[11];
    const float* upd_b2  = (const float*)d_in[12];
    const float* head_w1 = (const float*)d_in[13];
    const float* head_b1 = (const float*)d_in[14];
    const float* head_w2 = (const float*)d_in[15];
    const float* head_b2 = (const float*)d_in[16];

    // Workspace carve (256B aligned)
    char* ws = (char*)d_ws;
    size_t off = 0;
    auto carve = [&](size_t bytes) -> char* {
        char* p = ws + off;
        off = (off + bytes + 255) & ~(size_t)255;
        return p;
    };
    float*  h    = (float*) carve((size_t)N_NODES * DIM_H * 4);
    __bf16* hbuf = (__bf16*)carve((size_t)N_NODES * DIM_H * 2);
    float*  agg  = (float*) carve((size_t)N_NODES * DIM_H * 4);
    float*  cnt  = (float*) carve((size_t)N_NODES * 4);
    __bf16* w1p  = (__bf16*)carve((size_t)N_LAYERS * DIM_H * KP1 * 2);
    __bf16* w2p  = (__bf16*)carve((size_t)N_LAYERS * DIM_H * DIM_H * 2);
    __bf16* u1p  = (__bf16*)carve((size_t)N_LAYERS * DIM_H * KU * 2);
    __bf16* u2p  = (__bf16*)carve((size_t)N_LAYERS * DIM_H * DIM_H * 2);
    __bf16* hw1p = (__bf16*)carve((size_t)DIM_H * DIM_H * 2);

    const int TB = 256;
    auto nb = [](int n, int b) { return (n + b - 1) / b; };

    // 1) Pack all GEMM weights to bf16, transposed + K-padded
    pack_wT<<<nb(N_LAYERS * DIM_H * KP1,  TB), TB, 0, stream>>>(msg_w1, w1p, N_LAYERS, DIM_H + DIM_ED, DIM_H, KP1);
    pack_wT<<<nb(N_LAYERS * DIM_H * DIM_H,TB), TB, 0, stream>>>(msg_w2, w2p, N_LAYERS, DIM_H, DIM_H, DIM_H);
    pack_wT<<<nb(N_LAYERS * DIM_H * KU,   TB), TB, 0, stream>>>(upd_w1, u1p, N_LAYERS, KU, DIM_H, KU);
    pack_wT<<<nb(N_LAYERS * DIM_H * DIM_H,TB), TB, 0, stream>>>(upd_w2, u2p, N_LAYERS, DIM_H, DIM_H, DIM_H);
    pack_wT<<<nb(DIM_H * DIM_H,           TB), TB, 0, stream>>>(head_w1, hw1p, 1, DIM_H, DIM_H, DIM_H);

    // 2) In-degree counts (mean denominator)
    zero_f32 <<<nb(N_NODES, TB), TB, 0, stream>>>(cnt, N_NODES);
    count_deg<<<nb(N_EDGES, TB), TB, 0, stream>>>(ei, cnt, N_EDGES);

    // 3) Input projection
    lin_in_kernel<<<nb(N_NODES * DIM_H, TB), TB, 0, stream>>>(x, lin_w, lin_b, h, hbuf);

    // 4) Message-passing layers
    for (int l = 0; l < N_LAYERS; l++) {
        zero_f32x4<<<nb(N_NODES * DIM_H / 4, TB), TB, 0, stream>>>((f32x4*)agg, N_NODES * DIM_H / 4);
        msg_kernel<<<N_EDGES / 16, TB, 0, stream>>>(
            hbuf, ei, ea,
            w1p + (size_t)l * DIM_H * KP1,   msg_b1 + l * DIM_H,
            w2p + (size_t)l * DIM_H * DIM_H, msg_b2 + l * DIM_H,
            agg);
        upd_kernel<<<N_NODES / 16, TB, 0, stream>>>(
            h, hbuf, agg, cnt,
            u1p + (size_t)l * DIM_H * KU,    upd_b1 + l * DIM_H,
            u2p + (size_t)l * DIM_H * DIM_H, upd_b2 + l * DIM_H);
    }

    // 5) Head
    head_kernel<<<N_NODES / 16, TB, 0, stream>>>(hbuf, hw1p, head_b1,
                                                 head_w2, head_b2, (float*)d_out);
}
